// VoiceModel_5720896438806
// MI455X (gfx1250) — matrix operand order repaired and checked by reading, not compiled, not run on hardware
//
#include <hip/hip_runtime.h>
#include <math.h>

// ---------------- problem constants ----------------
#define B_    32
#define S_    128
#define T_    64
#define E_    1024
#define H_    1024
#define L_    2
#define VIN_  32000
#define VOUT_ 32000
#define G4H_  (4 * H_)

// ---------------- WMMA fragment types ----------------
typedef __attribute__((ext_vector_type(16))) __bf16 bf16x16;
typedef __attribute__((ext_vector_type(8)))  __bf16 bf16x8;
typedef __attribute__((ext_vector_type(8)))  float  f32x8;

// ---------------- CDNA5 async copy (global -> LDS), ASYNCcnt tracked ----------
static __device__ __forceinline__ void
async_b128_to_lds(unsigned int lds_addr, const void* gaddr) {
  asm volatile("global_load_async_to_lds_b128 %0, %1, off"
               :: "v"(lds_addr), "v"((unsigned long long)(uintptr_t)gaddr)
               : "memory");
}
static __device__ __forceinline__ void wait_async0() {
  asm volatile("s_wait_asynccnt 0" ::: "memory");
}

// B fragment: 32x16 bf16 tile of W^T; W row-major (N x K, ld = K).
// Lane n (0-15) holds column nBase+n, K kb..kb+15; lanes 16-31 hold K kb+16..31.
static __device__ __forceinline__ bf16x16
load_b_frag(const __bf16* W, int ld, int nBase, int kb, int lane) {
  const int col = nBase + (lane & 15);
  const int k0  = kb + ((lane & 16) ? 16 : 0);
  const __bf16* p = W + (size_t)col * ld + k0;
  __builtin_prefetch(p + 32, 0, 0);   // next K block -> global_prefetch_b8
  return *(const bf16x16*)(p);
}

// A fragment from a 32x32 bf16 LDS tile (row stride 32 elements).
// Lanes 0-15: row M=mBase+lane, K 0..15 ; lanes 16-31: K 16..31.
static __device__ __forceinline__ bf16x16
lds_a_frag(const __bf16* S, int mBase, int lane) {
  const int row = mBase + (lane & 15);
  const int k0  = (lane & 16) ? 16 : 0;
  const __bf16* p = S + row * 32 + k0;
  bf16x8 lo = *(const bf16x8*)(p);
  bf16x8 hi = *(const bf16x8*)(p + 8);
  bf16x16 a;
#pragma unroll
  for (int i = 0; i < 8; ++i) { a[i] = lo[i]; a[i + 8] = hi[i]; }
  return a;
}

// Stage one 32x32 bf16 K-block of the activation matrix into an LDS buffer.
// Executed by wave 0 only: lane l copies row l (64 bytes) with 4 async b128s.
static __device__ __forceinline__ void
stage_a_tile(__bf16* Sbuf, const __bf16* g_row /* row `lane` at col kv */, int lane) {
  const unsigned int lb = (unsigned int)(uintptr_t)(Sbuf + lane * 32);
  async_b128_to_lds(lb +  0, g_row +  0);
  async_b128_to_lds(lb + 16, g_row +  8);
  async_b128_to_lds(lb + 32, g_row + 16);
  async_b128_to_lds(lb + 48, g_row + 24);
}

// ---------------- LSTM gates GEMM ----------------
// gates(32 x 4H) = x(32 x E) @ Wih^T + h(32 x H) @ Whh^T + bih + bhh
// block = 256 (8 waves); each wave computes a 32x16 tile (2 accumulators,
// shared B fragment).  grid.x = 4H/128 = 32.
// A operand double-buffered in LDS via global_load_async_to_lds_b128.
__global__ void lstm_gates_gemm(const __bf16* __restrict__ x, int ldx,
                                const __bf16* __restrict__ h,
                                const __bf16* __restrict__ Wih,   // 4H x E
                                const __bf16* __restrict__ Whh,   // 4H x H
                                const float*  __restrict__ bih,
                                const float*  __restrict__ bhh,
                                float* __restrict__ gates)        // 32 x 4H
{
  __shared__ __align__(16) __bf16 sA[2][32 * 32];   // 2 x 2KB double buffer
  const int lane  = threadIdx.x & 31;
  const int wave  = threadIdx.x >> 5;
  const int nBase = blockIdx.x * 128 + wave * 16;
  const int KV    = E_ + H_;                        // virtual K = [x | h]

  // prologue: stage K-block 0 (from x)
  if (wave == 0) {
    stage_a_tile(&sA[0][0], x + (size_t)lane * ldx, lane);
    wait_async0();
  }
  __syncthreads();

  f32x8 acc0 = {}, acc1 = {};
  for (int kvb = 0; kvb < KV / 32; ++kvb) {
    const int kv  = kvb * 32;
    const int cur = kvb & 1;
    // stage next K-block while computing on current one
    if (wave == 0 && kv + 32 < KV) {
      const int kn = kv + 32;
      const __bf16* gp = (kn < E_) ? (x + (size_t)lane * ldx + kn)
                                   : (h + (size_t)lane * H_ + (kn - E_));
      stage_a_tile(&sA[cur ^ 1][0], gp, lane);
    }

    const __bf16* Wsrc = (kv < E_) ? Wih : Whh;
    const int     kw   = (kv < E_) ? kv : kv - E_;
    bf16x16 b  = load_b_frag(Wsrc, E_, nBase, kw, lane);
    bf16x16 a0 = lds_a_frag(&sA[cur][0], 0, lane);
    bf16x16 a1 = lds_a_frag(&sA[cur][0], 16, lane);
    acc0 = __builtin_amdgcn_wmma_f32_16x16x32_bf16(false, a0, false, b,
                                                   (short)0, acc0, false, false);
    acc1 = __builtin_amdgcn_wmma_f32_16x16x32_bf16(false, a1, false, b,
                                                   (short)0, acc1, false, false);

    if (wave == 0) wait_async0();
    __syncthreads();
  }

  const int n  = nBase + (lane & 15);
  const int r0 = (lane & 16) ? 8 : 0;
  const float bs = bih[n] + bhh[n];
#pragma unroll
  for (int v = 0; v < 8; ++v) {
    gates[(size_t)(r0 + v) * G4H_ + n]      = acc0[v] + bs;
    gates[(size_t)(16 + r0 + v) * G4H_ + n] = acc1[v] + bs;
  }
}

// ---------------- output projection GEMM ----------------
// out(32 x V) = h(32 x H) @ W^T + bias ; row stride ldOut (writes (b,t,:) slice)
// grid.x = V/128 = 250 ; block = 256.
__global__ void proj_gemm(const __bf16* __restrict__ h,     // 32 x H
                          const __bf16* __restrict__ W,     // V x H
                          const float*  __restrict__ bias,  // V
                          float* __restrict__ out, int ldOut)
{
  __shared__ __align__(16) __bf16 sA[2][32 * 32];
  const int lane  = threadIdx.x & 31;
  const int wave  = threadIdx.x >> 5;
  const int nBase = blockIdx.x * 128 + wave * 16;

  if (wave == 0) {
    stage_a_tile(&sA[0][0], h + (size_t)lane * H_, lane);
    wait_async0();
  }
  __syncthreads();

  f32x8 acc0 = {}, acc1 = {};
  for (int kvb = 0; kvb < H_ / 32; ++kvb) {
    const int kv  = kvb * 32;
    const int cur = kvb & 1;
    if (wave == 0 && kv + 32 < H_)
      stage_a_tile(&sA[cur ^ 1][0], h + (size_t)lane * H_ + kv + 32, lane);

    bf16x16 b  = load_b_frag(W, H_, nBase, kv, lane);
    bf16x16 a0 = lds_a_frag(&sA[cur][0], 0, lane);
    bf16x16 a1 = lds_a_frag(&sA[cur][0], 16, lane);
    acc0 = __builtin_amdgcn_wmma_f32_16x16x32_bf16(false, a0, false, b,
                                                   (short)0, acc0, false, false);
    acc1 = __builtin_amdgcn_wmma_f32_16x16x32_bf16(false, a1, false, b,
                                                   (short)0, acc1, false, false);

    if (wave == 0) wait_async0();
    __syncthreads();
  }

  const int n  = nBase + (lane & 15);
  const int r0 = (lane & 16) ? 8 : 0;
  const float bs = bias[n];
#pragma unroll
  for (int v = 0; v < 8; ++v) {
    out[(size_t)(r0 + v) * ldOut + n]      = acc0[v] + bs;
    out[(size_t)(16 + r0 + v) * ldOut + n] = acc1[v] + bs;
  }
}

// ---------------- elementwise LSTM cell ----------------
__global__ void lstm_cell(const float* __restrict__ gates,  // 32 x 4H
                          float*  __restrict__ c,           // 32 x H (fp32)
                          __bf16* __restrict__ h_state,     // 32 x H (bf16)
                          __bf16* __restrict__ h2,          // optional 2nd dest
                          int ld2)
{
  const int idx = blockIdx.x * blockDim.x + threadIdx.x;
  if (idx >= B_ * H_) return;
  const int b = idx >> 10;
  const int u = idx & (H_ - 1);
  const float* g = gates + (size_t)b * G4H_;
  const float gi = g[u];
  const float gf = g[H_ + u];
  const float gg = g[2 * H_ + u];
  const float go = g[3 * H_ + u];
  const float si = 1.0f / (1.0f + __expf(-gi));
  const float sf = 1.0f / (1.0f + __expf(-gf));
  const float so = 1.0f / (1.0f + __expf(-go));
  const float cn = sf * c[idx] + si * tanhf(gg);
  const float hn = so * tanhf(cn);
  c[idx] = cn;
  const __bf16 hb = (__bf16)hn;
  h_state[idx] = hb;
  if (h2) h2[(size_t)b * ld2 + u] = hb;
}

// ---------------- small utility kernels ----------------
__global__ void cast_f32_bf16(const float* __restrict__ in, __bf16* __restrict__ out, int n) {
  int i = blockIdx.x * blockDim.x + threadIdx.x;
  if (i < n) out[i] = (__bf16)in[i];
}
__global__ void zero_f32(float* p, int n) {
  int i = blockIdx.x * blockDim.x + threadIdx.x;
  if (i < n) p[i] = 0.0f;
}
__global__ void zero_bf16(__bf16* p, int n) {
  int i = blockIdx.x * blockDim.x + threadIdx.x;
  if (i < n) p[i] = (__bf16)0.0f;
}
__global__ void embed_tokens(const int* __restrict__ tok,
                             const __bf16* __restrict__ emb,
                             __bf16* __restrict__ dst, int nrows)
{
  int idx = blockIdx.x * blockDim.x + threadIdx.x;
  if (idx >= nrows * E_) return;
  const int r = idx >> 10;
  const int e = idx & (E_ - 1);
  dst[idx] = emb[(size_t)tok[r] * E_ + e];
}
__global__ void init_tok(int* __restrict__ next_tok, const int* __restrict__ target) {
  int b = threadIdx.x;
  if (b < B_) next_tok[b] = target[b * T_];   // target[:,0]
}
__global__ void zero_t0(float* __restrict__ out) {
  int idx = blockIdx.x * blockDim.x + threadIdx.x;
  if (idx >= B_ * VOUT_) return;
  const int b = idx / VOUT_;
  const int v = idx - b * VOUT_;
  out[(size_t)b * T_ * VOUT_ + v] = 0.0f;
}
__global__ void argmax_select(const float* __restrict__ out_t, int ldRow,
                              const int* __restrict__ target, int t,
                              const int* __restrict__ tf_mask,
                              int* __restrict__ next_tok)
{
  const int b = blockIdx.x;
  const float* row = out_t + (size_t)b * ldRow;
  float best = -INFINITY; int bi = 0;
  for (int v = threadIdx.x; v < VOUT_; v += blockDim.x) {
    float xv = row[v];
    if (xv > best || (xv == best && v < bi)) { best = xv; bi = v; }
  }
  __shared__ float sv[256];
  __shared__ int   si[256];
  sv[threadIdx.x] = best; si[threadIdx.x] = bi;
  __syncthreads();
  for (int s = 128; s > 0; s >>= 1) {
    if (threadIdx.x < s) {
      float ov = sv[threadIdx.x + s]; int oi = si[threadIdx.x + s];
      if (ov > sv[threadIdx.x] || (ov == sv[threadIdx.x] && oi < si[threadIdx.x])) {
        sv[threadIdx.x] = ov; si[threadIdx.x] = oi;
      }
    }
    __syncthreads();
  }
  if (threadIdx.x == 0)
    next_tok[b] = (tf_mask[t] > 0) ? target[b * T_ + t] : si[0];
}

// ---------------- host orchestration ----------------
extern "C" void kernel_launch(void* const* d_in, const int* in_sizes, int n_in,
                              void* d_out, int out_size, void* d_ws, size_t ws_size,
                              hipStream_t stream)
{
  const int*   input_tokens = (const int*)  d_in[0];   // (B,S)
  const int*   target       = (const int*)  d_in[1];   // (B,T)
  const int*   tf_mask      = (const int*)  d_in[2];   // (T,)
  const float* in_emb       = (const float*)d_in[3];   // (VIN,E)
  const float* out_emb      = (const float*)d_in[4];   // (VOUT,E)
  const float* enc_Wih      = (const float*)d_in[5];   // (L,4H,E)
  const float* enc_Whh      = (const float*)d_in[6];   // (L,4H,H)
  const float* enc_bih      = (const float*)d_in[7];   // (L,4H)
  const float* enc_bhh      = (const float*)d_in[8];
  const float* dec_Wih      = (const float*)d_in[9];
  const float* dec_Whh      = (const float*)d_in[10];
  const float* dec_bih      = (const float*)d_in[11];
  const float* dec_bhh      = (const float*)d_in[12];
  const float* W_out        = (const float*)d_in[13];  // (VOUT,H)
  const float* b_out        = (const float*)d_in[14];  // (VOUT,)
  float* out = (float*)d_out;                           // (B,T,VOUT)

  char* ws = (char*)d_ws;
  size_t off = 0;
  auto alloc = [&](size_t bytes) -> void* {
    void* p = ws + off;
    off = (off + bytes + 255) & ~(size_t)255;
    return p;
  };
  __bf16* in_emb_bf  = (__bf16*)alloc((size_t)VIN_  * E_ * 2);
  __bf16* out_emb_bf = (__bf16*)alloc((size_t)VOUT_ * E_ * 2);
  __bf16* encWih_bf  = (__bf16*)alloc((size_t)L_ * G4H_ * E_ * 2);
  __bf16* encWhh_bf  = (__bf16*)alloc((size_t)L_ * G4H_ * H_ * 2);
  __bf16* decWih_bf  = (__bf16*)alloc((size_t)L_ * G4H_ * E_ * 2);
  __bf16* decWhh_bf  = (__bf16*)alloc((size_t)L_ * G4H_ * H_ * 2);
  __bf16* Wout_bf    = (__bf16*)alloc((size_t)VOUT_ * H_ * 2);
  __bf16* seqA       = (__bf16*)alloc((size_t)B_ * S_ * E_ * 2);
  __bf16* seqB       = (__bf16*)alloc((size_t)B_ * S_ * E_ * 2);
  __bf16* h_state    = (__bf16*)alloc((size_t)L_ * B_ * H_ * 2);
  float*  c_state    = (float*) alloc((size_t)L_ * B_ * H_ * 4);
  float*  gates      = (float*) alloc((size_t)B_ * G4H_ * 4);
  __bf16* dec_x      = (__bf16*)alloc((size_t)B_ * E_ * 2);
  __bf16* dec_x2     = (__bf16*)alloc((size_t)B_ * H_ * 2);
  int*    next_tok   = (int*)   alloc((size_t)B_ * 4);
  (void)ws_size; (void)n_in; (void)in_sizes; (void)out_size;

  auto castN = [&](const float* src, __bf16* dst, size_t n) {
    cast_f32_bf16<<<(unsigned)((n + 255) / 256), 256, 0, stream>>>(src, dst, (int)n);
  };
  castN(in_emb,  in_emb_bf,  (size_t)VIN_  * E_);
  castN(out_emb, out_emb_bf, (size_t)VOUT_ * E_);
  castN(enc_Wih, encWih_bf,  (size_t)L_ * G4H_ * E_);
  castN(enc_Whh, encWhh_bf,  (size_t)L_ * G4H_ * H_);
  castN(dec_Wih, decWih_bf,  (size_t)L_ * G4H_ * E_);
  castN(dec_Whh, decWhh_bf,  (size_t)L_ * G4H_ * H_);
  castN(W_out,   Wout_bf,    (size_t)VOUT_ * H_);

  // ---- encoder ----
  embed_tokens<<<(B_ * S_ * E_) / 256, 256, 0, stream>>>(input_tokens, in_emb_bf, seqA, B_ * S_);

  for (int l = 0; l < L_; ++l) {
    __bf16* h_l = h_state + (size_t)l * B_ * H_;
    float*  c_l = c_state + (size_t)l * B_ * H_;
    zero_bf16<<<(B_ * H_) / 256, 256, 0, stream>>>(h_l, B_ * H_);
    zero_f32 <<<(B_ * H_) / 256, 256, 0, stream>>>(c_l, B_ * H_);
    const __bf16* seq_in  = (l == 0) ? seqA : seqB;
    __bf16*       seq_out = (l == 0) ? seqB : seqA;
    const __bf16* Wih = encWih_bf + (size_t)l * G4H_ * E_;
    const __bf16* Whh = encWhh_bf + (size_t)l * G4H_ * H_;
    const float*  bih = enc_bih + (size_t)l * G4H_;
    const float*  bhh = enc_bhh + (size_t)l * G4H_;
    for (int t = 0; t < S_; ++t) {
      lstm_gates_gemm<<<G4H_ / 128, 256, 0, stream>>>(
          seq_in + (size_t)t * E_, S_ * E_, h_l, Wih, Whh, bih, bhh, gates);
      lstm_cell<<<(B_ * H_) / 256, 256, 0, stream>>>(
          gates, c_l, h_l, seq_out + (size_t)t * E_, S_ * E_);
    }
  }

  // ---- decoder ----
  zero_t0<<<(B_ * VOUT_ + 255) / 256, 256, 0, stream>>>(out);
  init_tok<<<1, 32, 0, stream>>>(next_tok, target);

  __bf16* h0 = h_state;
  __bf16* h1 = h_state + (size_t)B_ * H_;
  float*  c0 = c_state;
  float*  c1 = c_state + (size_t)B_ * H_;

  for (int t = 1; t < T_; ++t) {
    embed_tokens<<<(B_ * E_) / 256, 256, 0, stream>>>(next_tok, out_emb_bf, dec_x, B_);
    // layer 0
    lstm_gates_gemm<<<G4H_ / 128, 256, 0, stream>>>(
        dec_x, E_, h0, decWih_bf, decWhh_bf, dec_bih, dec_bhh, gates);
    lstm_cell<<<(B_ * H_) / 256, 256, 0, stream>>>(gates, c0, h0, dec_x2, H_);
    // layer 1
    lstm_gates_gemm<<<G4H_ / 128, 256, 0, stream>>>(
        dec_x2, H_, h1,
        decWih_bf + (size_t)G4H_ * E_, decWhh_bf + (size_t)G4H_ * H_,
        dec_bih + G4H_, dec_bhh + G4H_, gates);
    lstm_cell<<<(B_ * H_) / 256, 256, 0, stream>>>(gates, c1, h1, (__bf16*)nullptr, 0);
    // projection into out[:, t, :]
    proj_gemm<<<VOUT_ / 128, 256, 0, stream>>>(
        h1, Wout_bf, b_out, out + (size_t)t * VOUT_, T_ * VOUT_);
    // next token (teacher forcing / greedy)
    argmax_select<<<B_, 256, 0, stream>>>(
        out + (size_t)t * VOUT_, T_ * VOUT_, target, t, tf_mask, next_tok);
  }
}
